// Performer_5720896438678
// MI455X (gfx1250) — compile-verified, matrix operand order checked by reading, not run on hardware
//
#include <hip/hip_runtime.h>
#include <math.h>

// ---------------------------------------------------------------------------
// Performer block for MI455X (gfx1250, wave32, WMMA).
// All GEMMs use v_wmma_f32_16x16x32_bf16 (bf16 inputs, f32 accumulate).
// bf16 (not f16) because FAVOR+ features exp(wtx - |k|^2/2) ~ 1e-13 underflow
// f16 but are representable in bf16's f32-range exponent.
// Each wave computes a 32x32 C block (2x2 WMMA tiles) so each K-step does
// 4 wmma per 8 b128 loads (128 B/wmma) instead of 1 per 4 (256 B/wmma).
// ---------------------------------------------------------------------------

typedef __bf16 bf16_t;
typedef __attribute__((ext_vector_type(16))) __bf16 v16bf;
typedef __attribute__((ext_vector_type(8)))  float  v8f;
typedef __attribute__((ext_vector_type(4)))  float  f32x4;

union Frag16 { v16bf v; struct { f32x4 lo, hi; } p; };

static constexpr int  BB = 8;            // batch
static constexpr int  NN = 8192;         // sequence length
static constexpr int  DD = 384;          // dim == emb
static constexpr int  MF = 192;          // random features m
static constexpr long RR = (long)BB * NN;  // 65536 flattened rows

__device__ __forceinline__ void load_frag(Frag16& f, const bf16_t* p) {
  f.p.lo = *(const f32x4*)(p);
  f.p.hi = *(const f32x4*)(p + 16);
}

// ---------------------------------------------------------------------------
// Generic WMMA GEMM:  C[M,N] = A[M,K] @ Bt[N,K]^T   (bf16 -> f32)
// Block = 128 threads = 4 waves arranged 2x2 over a 64x64 block tile; each
// wave owns a 32x32 C block = 2x2 wmma tiles. Fragments follow the gfx1250
// 16-bit A/B VGPR layout: lane = half*16 + m ; v16 elements 0..7 = K in
// [half*8, half*8+8), elements 8..15 = same + 16 (two 16B-aligned b128s).
// Epilogue consumes f32 per the C layout: acc[r] = C(8*half + r, lane&15).
// Requires M % 64 == 0, N % 64 == 0, K % 32 == 0 (true for all uses here).
// ---------------------------------------------------------------------------
template <typename Epi>
__global__ __launch_bounds__(128) void gemm_wmma_bf16(
    const bf16_t* __restrict__ A, const bf16_t* __restrict__ Bt,
    int M, int N, int K, long strideA, long strideB, Epi epi)
{
  const int wave  = threadIdx.x >> 5;
  const int lane  = threadIdx.x & 31;
  const int batch = blockIdx.z;
  A  += (long)batch * strideA;
  Bt += (long)batch * strideB;
  const int mn   = lane & 15;               // A row within tile / C column
  const int hsel = lane >> 4;               // lane-half -> K sub-block select
  const int mbase = blockIdx.y * 64 + (wave >> 1) * 32;
  const int nbase = blockIdx.x * 64 + (wave & 1) * 32;
  const bf16_t* a0 = A  + (long)(mbase      + mn) * K + hsel * 8;
  const bf16_t* a1 = A  + (long)(mbase + 16 + mn) * K + hsel * 8;
  const bf16_t* b0 = Bt + (long)(nbase      + mn) * K + hsel * 8;
  const bf16_t* b1 = Bt + (long)(nbase + 16 + mn) * K + hsel * 8;
  v8f acc00 = {}, acc01 = {}, acc10 = {}, acc11 = {};
  for (int k0 = 0; k0 < K; k0 += 32) {
    Frag16 fa0, fa1, fb0, fb1;
    load_frag(fa0, a0 + k0);
    load_frag(fa1, a1 + k0);
    load_frag(fb0, b0 + k0);
    load_frag(fb1, b1 + k0);
    __builtin_prefetch((const void*)(a0 + k0 + 256), 0, 3);  // global_prefetch_b8
    __builtin_prefetch((const void*)(b0 + k0 + 256), 0, 3);
    acc00 = __builtin_amdgcn_wmma_f32_16x16x32_bf16(false, fa0.v, false, fb0.v, (short)0, acc00, false, false);
    acc01 = __builtin_amdgcn_wmma_f32_16x16x32_bf16(false, fa0.v, false, fb1.v, (short)0, acc01, false, false);
    acc10 = __builtin_amdgcn_wmma_f32_16x16x32_bf16(false, fa1.v, false, fb0.v, (short)0, acc10, false, false);
    acc11 = __builtin_amdgcn_wmma_f32_16x16x32_bf16(false, fa1.v, false, fb1.v, (short)0, acc11, false, false);
  }
#pragma unroll
  for (int r = 0; r < 8; ++r) {
    const int r0 = mbase + hsel * 8 + r;
    epi(acc00[r], r0,      nbase + mn,      batch);
    epi(acc01[r], r0,      nbase + 16 + mn, batch);
    epi(acc10[r], r0 + 16, nbase + mn,      batch);
    epi(acc11[r], r0 + 16, nbase + 16 + mn, batch);
  }
}

// ------------------------------ epilogues ----------------------------------
struct EpiKQV {   // split kqv; v kept f32 for residual; v^T bf16 for kptv GEMM
  const float* bias; bf16_t* k16; bf16_t* q16; float* v32; bf16_t* vT;
  __device__ void operator()(float acc, int row, int col, int) const {
    float val = acc + bias[col];
    if (col < DD) {
      k16[(long)row * DD + col] = (bf16_t)val;
    } else if (col < 2 * DD) {
      q16[(long)row * DD + (col - DD)] = (bf16_t)val;
    } else {
      int c = col - 2 * DD;
      int b = row >> 13, t = row & (NN - 1);
      v32[(long)row * DD + c] = val;
      vT[((long)b * DD + c) * NN + t] = (bf16_t)val;
    }
  }
};
struct EpiKP {    // kp = exp(wtx - |k|^2/2)/sqrt(m), stored transposed [b,m,t]
  const float* xd; bf16_t* kpT;
  __device__ void operator()(float acc, int row, int col, int) const {
    float v = expf(acc - xd[row]) * 0.07216878364870322f;  // 1/sqrt(192)
    int b = row >> 13, t = row & (NN - 1);
    kpT[((long)b * MF + col) * NN + t] = (bf16_t)v;
  }
};
struct EpiQP {    // qp row-major [row, m]
  const float* xd; bf16_t* qp;
  __device__ void operator()(float acc, int row, int col, int) const {
    float v = expf(acc - xd[row]) * 0.07216878364870322f;
    qp[(long)row * MF + col] = (bf16_t)v;
  }
};
struct EpiKPTV {  // kptv[b, emb, m]
  bf16_t* kptv;
  __device__ void operator()(float acc, int row, int col, int batch) const {
    kptv[((long)batch * DD + row) * MF + col] = (bf16_t)acc;
  }
};
struct EpiAttn {  // y_attn = (qp @ kptv^T) / (D + eps)
  const float* dinv; bf16_t* yattn;
  __device__ void operator()(float acc, int row, int col, int batch) const {
    long gr = (long)batch * NN + row;
    yattn[gr * DD + col] = (bf16_t)(acc * dinv[gr]);
  }
};
struct EpiProj {  // y = v + y_attn @ proj_w + proj_b   (f32, into d_out)
  const float* pb; const float* v32; float* y;
  __device__ void operator()(float acc, int row, int col, int) const {
    long i = (long)row * DD + col;
    y[i] = v32[i] + acc + pb[col];
  }
};
struct EpiGelu {  // exact gelu(h2 @ w1 + b1)
  const float* b1; bf16_t* t1;
  __device__ void operator()(float acc, int row, int col, int) const {
    float u = acc + b1[col];
    float g = 0.5f * u * (1.0f + erff(u * 0.7071067811865475f));
    t1[(long)row * DD + col] = (bf16_t)g;
  }
};
struct EpiMlp2 {  // out = y + mlp  (in-place update of d_out)
  const float* b2; float* y;
  __device__ void operator()(float acc, int row, int col, int) const {
    long i = (long)row * DD + col;
    y[i] = y[i] + acc + b2[col];
  }
};

// --------------------------- elementwise kernels ---------------------------
__global__ __launch_bounds__(256) void ln_bf16_kernel(
    const float* __restrict__ x, const float* __restrict__ g,
    const float* __restrict__ b, bf16_t* __restrict__ out, long nrows)
{
  const int wave = threadIdx.x >> 5, lane = threadIdx.x & 31;
  const long row = (long)blockIdx.x * 8 + wave;
  if (row >= nrows) return;
  const float* xr = x + row * DD;
  float vals[DD / 32];
  float s = 0.f;
#pragma unroll
  for (int i = 0; i < DD / 32; ++i) { vals[i] = xr[lane + 32 * i]; s += vals[i]; }
#pragma unroll
  for (int o = 16; o > 0; o >>= 1) s += __shfl_xor(s, o, 32);
  const float mu = s * (1.0f / DD);
  float vs = 0.f;
#pragma unroll
  for (int i = 0; i < DD / 32; ++i) { float d = vals[i] - mu; vs += d * d; }
#pragma unroll
  for (int o = 16; o > 0; o >>= 1) vs += __shfl_xor(vs, o, 32);
  const float rstd = rsqrtf(vs * (1.0f / DD) + 1e-5f);
#pragma unroll
  for (int i = 0; i < DD / 32; ++i) {
    int c = lane + 32 * i;
    out[row * DD + c] = (bf16_t)((vals[i] - mu) * rstd * g[c] + b[c]);
  }
}

__global__ __launch_bounds__(256) void rowsq_kernel(
    const bf16_t* __restrict__ a, float* __restrict__ xd, long nrows)
{
  const int wave = threadIdx.x >> 5, lane = threadIdx.x & 31;
  const long row = (long)blockIdx.x * 8 + wave;
  if (row >= nrows) return;
  const bf16_t* r = a + row * DD;
  float s = 0.f;
#pragma unroll
  for (int i = 0; i < DD / 32; ++i) { float v = (float)r[lane + 32 * i]; s += v * v; }
#pragma unroll
  for (int o = 16; o > 0; o >>= 1) s += __shfl_xor(s, o, 32);
  if (lane == 0) xd[row] = 0.5f * s;
}

__global__ __launch_bounds__(256) void ksum_kernel(
    const bf16_t* __restrict__ kpT, float* __restrict__ ksum)
{
  const int wave = threadIdx.x >> 5, lane = threadIdx.x & 31;
  const int rowid = blockIdx.x * 8 + wave;   // b*MF + m
  if (rowid >= BB * MF) return;
  const bf16_t* p = kpT + (long)rowid * NN;
  float s = 0.f;
  for (int i = lane * 8; i < NN; i += 32 * 8) {
    f32x4 f = *(const f32x4*)(p + i);
    const bf16_t* h = (const bf16_t*)&f;
#pragma unroll
    for (int j = 0; j < 8; ++j) s += (float)h[j];
  }
#pragma unroll
  for (int o = 16; o > 0; o >>= 1) s += __shfl_xor(s, o, 32);
  if (lane == 0) ksum[rowid] = s;
}

__global__ __launch_bounds__(256) void dinv_kernel(
    const bf16_t* __restrict__ qp, const float* __restrict__ ksum,
    float* __restrict__ dinv)
{
  const int wave = threadIdx.x >> 5, lane = threadIdx.x & 31;
  const long row = (long)blockIdx.x * 8 + wave;
  if (row >= RR) return;
  const int b = (int)(row >> 13);
  const bf16_t* q = qp + row * MF;
  const float* ks = ksum + (long)b * MF;
  float s = 0.f;
#pragma unroll
  for (int i = 0; i < MF / 32; ++i) { int c = lane + 32 * i; s += (float)q[c] * ks[c]; }
#pragma unroll
  for (int o = 16; o > 0; o >>= 1) s += __shfl_xor(s, o, 32);
  if (lane == 0) dinv[row] = 1.0f / (s + 1e-8f);
}

__global__ __launch_bounds__(256) void cast_kernel(
    const float* __restrict__ W, bf16_t* __restrict__ o, long n)
{
  long i = (long)blockIdx.x * 256 + threadIdx.x;
  if (i < n) o[i] = (bf16_t)W[i];
}

__global__ __launch_bounds__(256) void castT_kernel(
    const float* __restrict__ W, bf16_t* __restrict__ o, int in, int out)
{
  long i = (long)blockIdx.x * 256 + threadIdx.x;
  if (i >= (long)in * out) return;
  int r = (int)(i / out), c = (int)(i % out);
  o[(long)c * in + r] = (bf16_t)W[i];   // [in,out] f32 -> [out,in] bf16
}

// ---------------------------------------------------------------------------
extern "C" void kernel_launch(void* const* d_in, const int* in_sizes, int n_in,
                              void* d_out, int out_size, void* d_ws, size_t ws_size,
                              hipStream_t stream)
{
  (void)in_sizes; (void)n_in; (void)out_size; (void)ws_size;
  const float* x      = (const float*)d_in[0];
  const float* w      = (const float*)d_in[1];
  const float* kqv_w  = (const float*)d_in[2];
  const float* kqv_b  = (const float*)d_in[3];
  const float* proj_w = (const float*)d_in[4];
  const float* proj_b = (const float*)d_in[5];
  const float* ln1_g  = (const float*)d_in[6];
  const float* ln1_b  = (const float*)d_in[7];
  const float* ln2_g  = (const float*)d_in[8];
  const float* ln2_b  = (const float*)d_in[9];
  const float* mlp_w1 = (const float*)d_in[10];
  const float* mlp_b1 = (const float*)d_in[11];
  const float* mlp_w2 = (const float*)d_in[12];
  const float* mlp_b2 = (const float*)d_in[13];
  float* y_out = (float*)d_out;   // holds y after proj, updated in-place by mlp2

  // ---- workspace carve (deterministic static layout; ~356 MB) ----
  char* p = (char*)d_ws;
  auto carve = [&](size_t bytes) -> char* {
    char* q = p; p += (bytes + 255) & ~(size_t)255; return q;
  };
  bf16_t* hA    = (bf16_t*)carve(RR * DD * 2);        // LN1(x); reused as yattn
  bf16_t* k16   = (bf16_t*)carve(RR * DD * 2);        // reused as h2
  bf16_t* q16   = (bf16_t*)carve(RR * DD * 2);        // reused as t1
  float*  v32   = (float* )carve(RR * DD * 4);
  bf16_t* vT    = (bf16_t*)carve(RR * DD * 2);        // [b, emb, n]
  bf16_t* kpT   = (bf16_t*)carve((long)BB * MF * NN * 2);  // [b, m, n]
  bf16_t* qp    = (bf16_t*)carve(RR * MF * 2);        // [row, m]
  bf16_t* kptv  = (bf16_t*)carve((long)BB * DD * MF * 2);  // [b, emb, m]
  float*  ksum  = (float* )carve((long)BB * MF * 4);
  float*  xd_k  = (float* )carve(RR * 4);
  float*  xd_q  = (float* )carve(RR * 4);
  float*  dinv  = (float* )carve(RR * 4);
  bf16_t* kqvWt = (bf16_t*)carve((long)3 * DD * DD * 2);   // [1152, 384]
  bf16_t* w16   = (bf16_t*)carve((long)MF * DD * 2);       // [192, 384] (already [N,K])
  bf16_t* projWt= (bf16_t*)carve((long)DD * DD * 2);
  bf16_t* m1Wt  = (bf16_t*)carve((long)DD * DD * 2);
  bf16_t* m2Wt  = (bf16_t*)carve((long)DD * DD * 2);
  bf16_t* yattn = hA;    // alias: hA dead after kqv GEMM
  bf16_t* h2    = k16;   // alias: k16 dead after wtx_k GEMM
  bf16_t* t1    = q16;   // alias: q16 dead after wtx_q GEMM

  // ---- weight prep (cast / transpose-cast to bf16 [N,K] layouts) ----
  cast_kernel <<<((long)MF * DD + 255) / 256, 256, 0, stream>>>(w, w16, (long)MF * DD);
  castT_kernel<<<((long)DD * 3 * DD + 255) / 256, 256, 0, stream>>>(kqv_w, kqvWt, DD, 3 * DD);
  castT_kernel<<<((long)DD * DD + 255) / 256, 256, 0, stream>>>(proj_w, projWt, DD, DD);
  castT_kernel<<<((long)DD * DD + 255) / 256, 256, 0, stream>>>(mlp_w1, m1Wt, DD, DD);
  castT_kernel<<<((long)DD * DD + 255) / 256, 256, 0, stream>>>(mlp_w2, m2Wt, DD, DD);

  const dim3 blk(128);
  // 1) h = LN1(x)
  ln_bf16_kernel<<<RR / 8, 256, 0, stream>>>(x, ln1_g, ln1_b, hA, RR);
  // 2) kqv = h @ kqv_w + b  ->  k16, q16, v32, vT
  gemm_wmma_bf16<<<dim3(3 * DD / 64, RR / 64, 1), blk, 0, stream>>>(
      hA, kqvWt, (int)RR, 3 * DD, DD, 0L, 0L,
      EpiKQV{kqv_b, k16, q16, v32, vT});
  // 3) xd = 0.5*|k|^2 , 0.5*|q|^2
  rowsq_kernel<<<RR / 8, 256, 0, stream>>>(k16, xd_k, RR);
  rowsq_kernel<<<RR / 8, 256, 0, stream>>>(q16, xd_q, RR);
  // 4) kp = exp(k @ w^T - xd_k)/sqrt(m) (transposed store) ; qp likewise
  gemm_wmma_bf16<<<dim3(MF / 64, RR / 64, 1), blk, 0, stream>>>(
      k16, w16, (int)RR, MF, DD, 0L, 0L, EpiKP{xd_k, kpT});
  gemm_wmma_bf16<<<dim3(MF / 64, RR / 64, 1), blk, 0, stream>>>(
      q16, w16, (int)RR, MF, DD, 0L, 0L, EpiQP{xd_q, qp});
  // 5) ksum[b,m] = sum_t kp ;  D = qp . ksum ;  dinv = 1/(D+eps)
  ksum_kernel<<<(BB * MF) / 8, 256, 0, stream>>>(kpT, ksum);
  dinv_kernel<<<RR / 8, 256, 0, stream>>>(qp, ksum, dinv);
  // 6) kptv[b] = v^T @ kp   (batched, K = 8192)
  gemm_wmma_bf16<<<dim3(MF / 64, DD / 64, BB), blk, 0, stream>>>(
      vT, kpT, DD, MF, NN, (long)DD * NN, (long)MF * NN, EpiKPTV{kptv});
  // 7) y_attn = (qp @ kptv^T) * dinv   (batched)
  gemm_wmma_bf16<<<dim3(DD / 64, NN / 64, BB), blk, 0, stream>>>(
      qp, kptv, NN, DD, MF, (long)NN * MF, (long)DD * MF, EpiAttn{dinv, yattn});
  // 8) y = v + y_attn @ proj_w + proj_b   (f32 into d_out)
  gemm_wmma_bf16<<<dim3(DD / 64, RR / 64, 1), blk, 0, stream>>>(
      yattn, projWt, (int)RR, DD, DD, 0L, 0L, EpiProj{proj_b, v32, y_out});
  // 9) h2 = LN2(y)
  ln_bf16_kernel<<<RR / 8, 256, 0, stream>>>(y_out, ln2_g, ln2_b, h2, RR);
  // 10) t1 = gelu(h2 @ mlp_w1 + b1)
  gemm_wmma_bf16<<<dim3(DD / 64, RR / 64, 1), blk, 0, stream>>>(
      h2, m1Wt, (int)RR, DD, DD, 0L, 0L, EpiGelu{mlp_b1, t1});
  // 11) out = y + t1 @ mlp_w2 + b2
  gemm_wmma_bf16<<<dim3(DD / 64, RR / 64, 1), blk, 0, stream>>>(
      t1, m2Wt, (int)RR, DD, DD, 0L, 0L, EpiMlp2{mlp_b2, y_out});
}
